// MySpikeGPT_53463752901303
// MI455X (gfx1250) — compile-verified, matrix operand order checked by reading, not run on hardware
//
#include <hip/hip_runtime.h>
#include <hip/hip_fp16.h>
#include <hip/hip_bf16.h>

// ---------------- model dims ----------------
#define TT 4
#define BB 2
#define SS 1024
#define DD 768
#define HH 12
#define HDD 64
#define FF 3072
#define LL 4
#define VV 8000
#define EPSN 1e-5f

// ---------------- WMMA types ----------------
typedef __attribute__((ext_vector_type(16))) _Float16 v16h;
typedef __attribute__((ext_vector_type(8)))  float    v8f;

union F16x16 { v16h v; uint4 u[2]; };

// ---------------- CDNA5 async global->LDS copy ----------------
typedef int gv4i __attribute__((vector_size(16)));

#if __has_builtin(__builtin_amdgcn_global_load_async_to_lds_b128)
#define HAVE_ASYNC_LDS 1
#define ASYNC_COPY16(g, l)                                                    \
    __builtin_amdgcn_global_load_async_to_lds_b128(                           \
        (__attribute__((address_space(1))) gv4i*)(g),                         \
        (__attribute__((address_space(3))) gv4i*)(l), 0, 0)
#else
#define HAVE_ASYNC_LDS 0
#define ASYNC_COPY16(g, l)                                                    \
    (*reinterpret_cast<uint4*>(l) = *reinterpret_cast<const uint4*>(g))
#endif

#if HAVE_ASYNC_LDS
#if __has_builtin(__builtin_amdgcn_s_wait_asynccnt)
#define ASYNC_WAIT() __builtin_amdgcn_s_wait_asynccnt(0)
#else
#define ASYNC_WAIT() asm volatile("s_wait_asynccnt 0" ::: "memory")
#endif
#else
#define ASYNC_WAIT() ((void)0)
#endif

// ---------------- GEMM tiling ----------------
#define BM 256
#define BN 128
#define BK 32
#define LDK 40            // 32 + 8 halves pad -> 80B rows, 16B aligned

// C[M,N] = A[M,K](f16 row-major) * Bt[N,K](f16, B transposed) (+bias[n])
// mode 0: store C.  mode 1: no store; per-b sum/sumsq into stats[2*b..]
// 8 waves; wave (wr,wc) owns 64x64 macro-tile = 4x4 WMMA tiles.
__global__ __launch_bounds__(256)
void gemm_wmma(const __half* __restrict__ A, const __half* __restrict__ Bt,
               const float* __restrict__ bias, float* __restrict__ C,
               float* __restrict__ stats, int M, int N, int K, int mode)
{
    __shared__ __align__(16) __half As[2][BM * LDK];   // 2 x 20KB
    __shared__ __align__(16) __half Bs[2][BN * LDK];   // 2 x 10KB

    const int tid  = threadIdx.x;
    const int lane = tid & 31;
    const int wv   = tid >> 5;      // 8 waves
    const int wr   = wv >> 1;       // 0..3 (M)
    const int wc   = wv & 1;        // 0..1 (N)
    const int m0   = blockIdx.x * BM;
    const int n0   = blockIdx.y * BN;
    const int hi   = (lane >> 4) & 1;
    const int l16  = lane & 15;

    v8f acc[4][4];
#pragma unroll
    for (int i = 0; i < 4; ++i)
#pragma unroll
        for (int j = 0; j < 4; ++j)
#pragma unroll
            for (int e = 0; e < 8; ++e) acc[i][j][e] = 0.0f;

    // stage one (A,B) K-slice into LDS buffer `buf` (async if available)
    auto load_tiles = [&](int kb, int buf) {
        // A: 256 rows x 32 halves = 1024 x 16B chunks -> 4 per thread
#pragma unroll
        for (int j = 0; j < 4; ++j) {
            int c = tid + 256 * j;
            int row = c >> 2, q = c & 3;
            ASYNC_COPY16(A + (size_t)(m0 + row) * K + kb + q * 8,
                         &As[buf][row * LDK + q * 8]);
        }
        // Bt: 128 rows x 32 halves = 512 x 16B chunks -> 2 per thread
#pragma unroll
        for (int j = 0; j < 2; ++j) {
            int c = tid + 256 * j;
            int row = c >> 2, q = c & 3;
            __half* l = &Bs[buf][row * LDK + q * 8];
            if (n0 + row < N) {
                ASYNC_COPY16(Bt + (size_t)(n0 + row) * K + kb + q * 8, l);
            } else {
                uint4 z; z.x = z.y = z.z = z.w = 0u;
                *reinterpret_cast<uint4*>(l) = z;
            }
        }
    };

    const int nk = K / BK;
    load_tiles(0, 0);
    ASYNC_WAIT();
    __syncthreads();

    for (int ik = 0; ik < nk; ++ik) {
        const int buf = ik & 1;
        if (ik + 1 < nk) load_tiles((ik + 1) * BK, buf ^ 1);  // prefetch next slice

        F16x16 af[4], bf[4];
#pragma unroll
        for (int mi = 0; mi < 4; ++mi) {
            int r = (wr * 64 + mi * 16 + l16) * LDK;
            af[mi].u[0] = *reinterpret_cast<const uint4*>(&As[buf][r + hi * 8]);
            af[mi].u[1] = *reinterpret_cast<const uint4*>(&As[buf][r + 16 + hi * 8]);
        }
#pragma unroll
        for (int ni = 0; ni < 4; ++ni) {
            int r = (wc * 64 + ni * 16 + l16) * LDK;
            bf[ni].u[0] = *reinterpret_cast<const uint4*>(&Bs[buf][r + hi * 8]);
            bf[ni].u[1] = *reinterpret_cast<const uint4*>(&Bs[buf][r + 16 + hi * 8]);
        }
#pragma unroll
        for (int mi = 0; mi < 4; ++mi)
#pragma unroll
            for (int ni = 0; ni < 4; ++ni)
                acc[mi][ni] = __builtin_amdgcn_wmma_f32_16x16x32_f16(
                    false, af[mi].v, false, bf[ni].v, (short)0, acc[mi][ni], false, false);

        if (ik + 1 < nk) ASYNC_WAIT();   // next slice landed in LDS
        __syncthreads();                 // everyone done reading `buf`
    }

    if (mode == 0) {
#pragma unroll
        for (int mi = 0; mi < 4; ++mi)
#pragma unroll
            for (int ni = 0; ni < 4; ++ni)
#pragma unroll
                for (int vr = 0; vr < 8; ++vr) {
                    int row = m0 + wr * 64 + mi * 16 + hi * 8 + vr;
                    int col = n0 + wc * 64 + ni * 16 + l16;
                    if (col < N) {
                        float x = acc[mi][ni][vr];
                        if (bias) x += bias[col];
                        C[(size_t)row * N + col] = x;
                    }
                }
    } else {
        float s1 = 0.f, s2 = 0.f;
#pragma unroll
        for (int mi = 0; mi < 4; ++mi)
#pragma unroll
            for (int ni = 0; ni < 4; ++ni)
#pragma unroll
                for (int vr = 0; vr < 8; ++vr) {
                    int col = n0 + wc * 64 + ni * 16 + l16;
                    if (col < N) {
                        float x = acc[mi][ni][vr];
                        if (bias) x += bias[col];
                        s1 += x; s2 += x * x;
                    }
                }
        __shared__ float r1[256], r2[256];
        r1[tid] = s1; r2[tid] = s2;
        __syncthreads();
        for (int s = 128; s > 0; s >>= 1) {
            if (tid < s) { r1[tid] += r1[tid + s]; r2[tid] += r2[tid + s]; }
            __syncthreads();
        }
        if (tid == 0) {
            int b = (m0 / SS) % BB;   // rows are (t*B+b)*S+s; tile inside one (t,b)
            atomicAdd(&stats[2 * b + 0], r1[0]);
            atomicAdd(&stats[2 * b + 1], r2[0]);
        }
    }
}

// ---------------- elementwise / small kernels ----------------
__global__ void zero_f(float* p, int n) {
    int i = blockIdx.x * blockDim.x + threadIdx.x;
    if (i < n) p[i] = 0.f;
}

// Wt[n*K+k] = (half)W[k*N+n]
__global__ void tr_half(const float* __restrict__ W, __half* __restrict__ Wt, int K, int N) {
    long i = (long)blockIdx.x * blockDim.x + threadIdx.x;
    if (i >= (long)K * N) return;
    int k = (int)(i / N), n = (int)(i % N);
    Wt[(size_t)n * K + k] = __float2half(W[i]);
}

__global__ void embed_kernel(const int* __restrict__ tok, const float* __restrict__ emb,
                             float* __restrict__ x0) {
    long i = (long)blockIdx.x * blockDim.x + threadIdx.x;
    if (i >= (long)BB * SS * DD) return;
    int d = (int)(i % DD);
    int s = (int)((i / DD) % SS);
    int b = (int)(i / ((long)SS * DD));
    int t = tok[b * SS + s];
    float je  = (float)((d >> 1) << 1);
    float ang = (float)s * __expf(-(je / (float)DD) * logf(10000.0f));
    float poe = ((d & 1) == 0) ? __sinf(ang) : __cosf(ang);
    x0[i] = emb[(size_t)t * DD + d] + poe;
}

// accumulate sum/sumsq per b over [T_, B, SF] layout; grid.y = b
__global__ void bn_accum(const float* __restrict__ x, float* __restrict__ st, long SF, int T_) {
    int b = blockIdx.y;
    long n = (long)T_ * SF;
    float s1 = 0.f, s2 = 0.f;
    for (long j = (long)blockIdx.x * blockDim.x + threadIdx.x; j < n;
         j += (long)gridDim.x * blockDim.x) {
        long t = j / SF, o = j - t * SF;
        float v = x[(t * BB + b) * SF + o];
        s1 += v; s2 += v * v;
    }
    __shared__ float r1[256], r2[256];
    int tid = threadIdx.x;
    r1[tid] = s1; r2[tid] = s2;
    __syncthreads();
    for (int s = 128; s > 0; s >>= 1) {
        if (tid < s) { r1[tid] += r1[tid + s]; r2[tid] += r2[tid + s]; }
        __syncthreads();
    }
    if (tid == 0) { atomicAdd(&st[2 * b], r1[0]); atomicAdd(&st[2 * b + 1], r2[0]); }
}

__global__ void bn_finalize(const float* __restrict__ st, float* __restrict__ ms, float count) {
    int b = threadIdx.x;
    if (b < BB) {
        float mean = st[2 * b] / count;
        float var  = st[2 * b + 1] / count - mean * mean;
        ms[2 * b]     = mean;
        ms[2 * b + 1] = rsqrtf(var + EPSN);
    }
}

__global__ void bn_apply(float* __restrict__ x, const float* __restrict__ ms, long SF, long total) {
    long i = (long)blockIdx.x * blockDim.x + threadIdx.x;
    if (i >= total) return;
    int b = (int)((i / SF) % BB);
    x[i] = (x[i] - ms[2 * b]) * ms[2 * b + 1];
}

__global__ void bn_apply_rep(const float* __restrict__ x0, const float* __restrict__ ms,
                             float* __restrict__ X) {
    long i = (long)blockIdx.x * blockDim.x + threadIdx.x;
    if (i >= (long)TT * BB * SS * DD) return;
    long e = i % ((long)BB * SS * DD);
    int  b = (int)(e / ((long)SS * DD));
    X[i] = (x0[e] - ms[2 * b]) * ms[2 * b + 1];
}

// multi-step IF neuron: x [TT, n], per-t stride n; spikes as f16 {0,1}
__global__ void if_node_k(const float* __restrict__ x, __half* __restrict__ sp,
                          const float* __restrict__ vin, float* __restrict__ vout, long n) {
    long i = (long)blockIdx.x * blockDim.x + threadIdx.x;
    if (i >= n) return;
    float v = vin ? vin[i] : 0.f;
#pragma unroll
    for (int t = 0; t < TT; ++t) {
        v += x[(long)t * n + i];
        float s = (v - 1.0f) >= 0.f ? 1.f : 0.f;
        sp[(long)t * n + i] = __float2half(s);
        v *= (1.f - s);
    }
    if (vout) vout[i] = v;
}

// qk[t*B+b][d] = sum_s Q[tb,s,d]*K[tb,s,d]
__global__ void qk_reduce(const __half* __restrict__ Q, const __half* __restrict__ Ks,
                          float* __restrict__ qk) {
    int tb = blockIdx.y;
    int d  = blockIdx.x * 256 + threadIdx.x;
    if (d >= DD) return;
    size_t base = (size_t)tb * SS * DD + d;
    float s = 0.f;
    for (int ss = 0; ss < SS; ++ss)
        s += __half2float(Q[base + (size_t)ss * DD]) * __half2float(Ks[base + (size_t)ss * DD]);
    qk[(size_t)tb * DD + d] = s;
}

// O[t,b,s,d] = Vs[t,b,s,d] * qks[t,b,d]
__global__ void vo_mul(const __half* __restrict__ Vs, const __half* __restrict__ qks,
                       __half* __restrict__ O) {
    long i = (long)blockIdx.x * blockDim.x + threadIdx.x;
    if (i >= (long)TT * BB * SS * DD) return;
    int d  = (int)(i % DD);
    int tb = (int)(i / ((long)SS * DD));
    float r = __half2float(Vs[i]) * __half2float(qks[(size_t)tb * DD + d]);
    O[i] = __float2half(r);
}

__global__ void add_hs(const __half* __restrict__ xs, const float* __restrict__ att,
                       float* __restrict__ x, long n) {
    long i = (long)blockIdx.x * blockDim.x + threadIdx.x;
    if (i < n) x[i] = __half2float(xs[i]) + att[i];
}

__global__ void add_inplace(float* __restrict__ x, const float* __restrict__ f, long n) {
    long i = (long)blockIdx.x * blockDim.x + threadIdx.x;
    if (i < n) x[i] += f[i];
}

// logits at s = S-1 only, BN-applied with precomputed stats
__global__ void last_logits(const __half* __restrict__ YS, const __half* __restrict__ WoutT,
                            const float* __restrict__ bout, const float* __restrict__ ms,
                            float* __restrict__ LROW) {
    int r = blockIdx.y;                 // t*B+b
    int n = blockIdx.x * 256 + threadIdx.x;
    if (n >= VV) return;
    int b = r % BB;
    const __half* ys = YS + ((size_t)r * SS + (SS - 1)) * DD;
    const __half* w  = WoutT + (size_t)n * DD;
    float s = bout[n];
    for (int k = 0; k < DD; ++k) s += __half2float(ys[k]) * __half2float(w[k]);
    LROW[(size_t)r * VV + n] = (s - ms[2 * b]) * ms[2 * b + 1];
}

__global__ void final_spike_mean(const float* __restrict__ LROW, float* __restrict__ MROW,
                                 float* __restrict__ msum) {
    long i = (long)blockIdx.x * blockDim.x + threadIdx.x;
    if (i >= (long)BB * VV) return;
    int b = (int)(i / VV);
    float v = 0.f, m = 0.f;
#pragma unroll
    for (int t = 0; t < TT; ++t) {
        v += LROW[(long)t * BB * VV + i];
        float s = (v - 1.0f) >= 0.f ? 1.f : 0.f;
        m += s;
        v *= (1.f - s);
    }
    m *= (1.0f / TT);
    MROW[i] = m;
    atomicAdd(&msum[b], m);
}

__global__ void final_norm(const float* __restrict__ MROW, const float* __restrict__ msum,
                           float* __restrict__ out) {
    long i = (long)blockIdx.x * blockDim.x + threadIdx.x;
    if (i >= (long)BB * VV) return;
    int b = (int)(i / VV);
    out[i] = MROW[i] / msum[b];
}

// ---------------- host orchestration ----------------
extern "C" void kernel_launch(void* const* d_in, const int* in_sizes, int n_in,
                              void* d_out, int out_size, void* d_ws, size_t ws_size,
                              hipStream_t stream) {
    const int*   tokens = (const int*)d_in[0];
    const float* emb    = (const float*)d_in[1];
    const float* wq     = (const float*)d_in[2];
    const float* wk     = (const float*)d_in[3];
    const float* wv     = (const float*)d_in[4];
    const float* wo     = (const float*)d_in[5];
    const float* w1     = (const float*)d_in[6];
    const float* b1     = (const float*)d_in[7];
    const float* w2     = (const float*)d_in[8];
    const float* b2     = (const float*)d_in[9];
    const float* wout   = (const float*)d_in[10];
    const float* bout   = (const float*)d_in[11];
    float*       out    = (float*)d_out;

    const long TBSD = (long)TT * BB * SS * DD;     // 6.29M
    const long BSD  = (long)BB * SS * DD;          // 1.57M
    const long BSF  = (long)BB * SS * FF;          // 6.29M
    const int  M    = TT * BB * SS;                // 8192

    // bump allocator over workspace (~441 MB total)
    char*  ws  = (char*)d_ws;
    size_t off = 0;
    auto alloc = [&](size_t bytes) -> void* {
        void* p = ws + off;
        off += (bytes + 255) & ~(size_t)255;
        return p;
    };

    float*  X    = (float*)alloc(TBSD * 4);
    __half* XS   = (__half*)alloc(TBSD * 2);
    float*  QF   = (float*)alloc(TBSD * 4);
    float*  VF   = (float*)alloc(TBSD * 4);
    float*  KF   = (float*)alloc(TBSD * 4);
    __half* QS   = (__half*)alloc(TBSD * 2);
    __half* VS   = (__half*)alloc(TBSD * 2);
    __half* KS   = (__half*)alloc(TBSD * 2);
    float*  VMEM = (float*)alloc(BSD * 4);
    float*  QK   = (float*)alloc((size_t)TT * BB * DD * 4);
    __half* QKS  = (__half*)alloc((size_t)TT * BB * DD * 2);
    __half* O    = (__half*)alloc(TBSD * 2);
    float*  ATT  = (float*)alloc(TBSD * 4);
    float*  G    = (float*)alloc((size_t)TT * BSF * 4);
    __half* GS   = (__half*)alloc((size_t)TT * BSF * 2);
    float*  Fb   = (float*)alloc(TBSD * 4);
    __half* wqT  = (__half*)alloc((size_t)LL * DD * DD * 2);
    __half* wkT  = (__half*)alloc((size_t)LL * DD * DD * 2);
    __half* wvT  = (__half*)alloc((size_t)LL * DD * DD * 2);
    __half* woT  = (__half*)alloc((size_t)LL * DD * DD * 2);
    __half* w1T  = (__half*)alloc((size_t)LL * DD * FF * 2);
    __half* w2T  = (__half*)alloc((size_t)LL * FF * DD * 2);
    __half* woutT= (__half*)alloc((size_t)VV * DD * 2);
    float*  LROW = (float*)alloc((size_t)TT * BB * VV * 4);
    float*  MROW = (float*)alloc((size_t)BB * VV * 4);
    float*  STATS= (float*)alloc(2 * BB * 4);
    float*  MS   = (float*)alloc(2 * BB * 4);
    float*  MSUM = (float*)alloc(BB * 4);
    (void)ws_size; (void)n_in; (void)in_sizes; (void)out_size;

    auto blocks = [](long n) { return (unsigned)((n + 255) / 256); };

    auto bn_stats = [&](const float* x, long SF, int T_, float count) {
        zero_f<<<1, 64, 0, stream>>>(STATS, 2 * BB);
        dim3 g(256, BB);
        bn_accum<<<g, 256, 0, stream>>>(x, STATS, SF, T_);
        bn_finalize<<<1, 32, 0, stream>>>(STATS, MS, count);
    };
    auto bn_do = [&](float* x, long SF, long total) {
        bn_apply<<<blocks(total), 256, 0, stream>>>(x, MS, SF, total);
    };
    auto gemm = [&](const __half* A, const __half* Bt, const float* bias, float* C, int N, int K) {
        dim3 grid(M / BM, (N + BN - 1) / BN);
        gemm_wmma<<<grid, 256, 0, stream>>>(A, Bt, bias, C, nullptr, M, N, K, 0);
    };
    auto gemm_stats = [&](const __half* A, const __half* Bt, const float* bias, int N, int K) {
        zero_f<<<1, 64, 0, stream>>>(STATS, 2 * BB);
        dim3 grid(M / BM, (N + BN - 1) / BN);
        gemm_wmma<<<grid, 256, 0, stream>>>(A, Bt, bias, nullptr, STATS, M, N, K, 1);
    };

    // ---- weight conversion (transposed, f16) ----
    for (int l = 0; l < LL; ++l) {
        long dd = (long)DD * DD, df = (long)DD * FF;
        tr_half<<<blocks(dd), 256, 0, stream>>>(wq + (size_t)l * dd, wqT + (size_t)l * dd, DD, DD);
        tr_half<<<blocks(dd), 256, 0, stream>>>(wk + (size_t)l * dd, wkT + (size_t)l * dd, DD, DD);
        tr_half<<<blocks(dd), 256, 0, stream>>>(wv + (size_t)l * dd, wvT + (size_t)l * dd, DD, DD);
        tr_half<<<blocks(dd), 256, 0, stream>>>(wo + (size_t)l * dd, woT + (size_t)l * dd, DD, DD);
        tr_half<<<blocks(df), 256, 0, stream>>>(w1 + (size_t)l * df, w1T + (size_t)l * df, DD, FF);
        tr_half<<<blocks(df), 256, 0, stream>>>(w2 + (size_t)l * df, w2T + (size_t)l * df, FF, DD);
    }
    tr_half<<<blocks((long)DD * VV), 256, 0, stream>>>(wout, woutT, DD, VV);

    // ---- embedding + positional + BN (broadcast over T) ----
    float* X0 = QF;   // temp reuse
    embed_kernel<<<blocks(BSD), 256, 0, stream>>>(tokens, emb, X0);
    bn_stats(X0, (long)SS * DD, 1, (float)((long)SS * DD));
    bn_apply_rep<<<blocks(TBSD), 256, 0, stream>>>(X0, MS, X);

    const float cntD = (float)((long)TT * SS * DD);
    const float cntF = (float)((long)TT * SS * FF);

    for (int l = 0; l < LL; ++l) {
        const __half* wqTl = wqT + (size_t)l * DD * DD;
        const __half* wkTl = wkT + (size_t)l * DD * DD;
        const __half* wvTl = wvT + (size_t)l * DD * DD;
        const __half* woTl = woT + (size_t)l * DD * DD;
        const __half* w1Tl = w1T + (size_t)l * DD * FF;
        const __half* w2Tl = w2T + (size_t)l * FF * DD;
        const float*  b1l  = b1 + (size_t)l * FF;
        const float*  b2l  = b2 + (size_t)l * DD;

        // ---- SDSA ----
        if_node_k<<<blocks(BSD), 256, 0, stream>>>(X, XS, nullptr, nullptr, BSD);      // xs
        gemm(XS, wqTl, nullptr, QF, DD, DD);
        gemm(XS, wvTl, nullptr, VF, DD, DD);
        gemm(XS, wkTl, nullptr, KF, DD, DD);
        bn_stats(QF, (long)SS * DD, TT, cntD); bn_do(QF, (long)SS * DD, TBSD);
        if_node_k<<<blocks(BSD), 256, 0, stream>>>(QF, QS, nullptr, VMEM, BSD);        // Qs, v
        bn_stats(VF, (long)SS * DD, TT, cntD); bn_do(VF, (long)SS * DD, TBSD);
        if_node_k<<<blocks(BSD), 256, 0, stream>>>(VF, VS, VMEM, VMEM, BSD);           // Vs, v
        bn_stats(KF, (long)SS * DD, TT, cntD); bn_do(KF, (long)SS * DD, TBSD);
        if_node_k<<<blocks(BSD), 256, 0, stream>>>(KF, KS, VMEM, VMEM, BSD);           // Ks
        {
            dim3 g((DD + 255) / 256, TT * BB);
            qk_reduce<<<g, 256, 0, stream>>>(QS, KS, QK);
        }
        if_node_k<<<blocks((long)BB * DD), 256, 0, stream>>>(QK, QKS, nullptr, nullptr,
                                                             (long)BB * DD);           // talking heads
        vo_mul<<<blocks(TBSD), 256, 0, stream>>>(VS, QKS, O);
        gemm(O, woTl, nullptr, ATT, DD, DD);
        bn_stats(ATT, (long)SS * DD, TT, cntD); bn_do(ATT, (long)SS * DD, TBSD);
        add_hs<<<blocks(TBSD), 256, 0, stream>>>(XS, ATT, X, TBSD);                    // h = xs + attn

        // ---- FFN ----
        if_node_k<<<blocks(BSD), 256, 0, stream>>>(X, XS, nullptr, nullptr, BSD);      // hs
        gemm(XS, w1Tl, b1l, G, FF, DD);
        bn_stats(G, (long)SS * FF, TT, cntF); bn_do(G, (long)SS * FF, (long)M * FF);
        if_node_k<<<blocks(BSF), 256, 0, stream>>>(G, GS, nullptr, nullptr, BSF);
        gemm(GS, w2Tl, b2l, Fb, DD, FF);
        bn_stats(Fb, (long)SS * DD, TT, cntD); bn_do(Fb, (long)SS * DD, TBSD);
        add_inplace<<<blocks(TBSD), 256, 0, stream>>>(X, Fb, TBSD);                    // x = h + ffn
    }

    // ---- head: ys = IF(x); logits BN via stats-only GEMM; output at s=S-1 ----
    if_node_k<<<blocks(BSD), 256, 0, stream>>>(X, XS, nullptr, nullptr, BSD);          // ys
    gemm_stats(XS, woutT, bout, VV, DD);
    bn_finalize<<<1, 32, 0, stream>>>(STATS, MS, (float)((long)TT * SS * VV));
    {
        dim3 g((VV + 255) / 256, TT * BB);
        last_logits<<<g, 256, 0, stream>>>(XS, woutT, bout, MS, LROW);
    }
    zero_f<<<1, 64, 0, stream>>>(MSUM, BB);
    final_spike_mean<<<blocks((long)BB * VV), 256, 0, stream>>>(LROW, MROW, MSUM);
    final_norm<<<blocks((long)BB * VV), 256, 0, stream>>>(MROW, MSUM, out);
}
